// NgramEmbedding_51359218925899
// MI455X (gfx1250) — matrix operand order repaired and checked by reading, not compiled, not run on hardware
//
#include <hip/hip_runtime.h>
#include <hip/hip_bf16.h>

// ---------------------------------------------------------------------------
// NgramEmbedding for MI455X (gfx1250, wave32, WMMA, async global->LDS)
//
//   out[b,s,h] = ( word_emb[ids[b,s],h]
//                + sum_idx  proj_idx[h,:] . emb_idx[hash_idx(b,s), :] ) / 5
//
// Stages (all on `stream`, serialized by stream order):
//   1) hash_kernel    : rolling-hash ngram ids (EOS check is a 2-token lookback)
//   2) cvt_proj_kernel: proj fp32 -> bf16 staging (row-major [idx][2048][512])
//   3) gather_kernel  : emb gather fp32 -> bf16 staging (row-major [idx][8192][512])
//   4) gemm_kernel    : fused 4x (8192x512x2048) bf16 WMMA GEMM,
//                       GLOBAL_LOAD_ASYNC_TO_LDS_B128 double-buffered panels,
//                       word_emb gather epilogue + 1/5 scale
// ---------------------------------------------------------------------------

#define B_      4
#define S_      2048
#define M_      (B_ * S_)        // 8192 rows
#define HID_    2048
#define EDIM_   512
#define NEMB_   4
#define EOS_    2
#define VOCAB_  32000

// GEMM tiling
#define BM_     256              // block tile M
#define BN_     128              // block tile N
#define BK_     32               // K step (one bf16 WMMA K)
#define LDSROW_ 40               // padded row stride in ushorts (80 B, 16B-aligned)
#define NITER_  (NEMB_ * (EDIM_ / BK_))   // 64 K-panel iterations

typedef __attribute__((ext_vector_type(16))) __bf16        v16bf;
typedef __attribute__((ext_vector_type(8)))  float         v8f;
typedef __attribute__((ext_vector_type(8)))  unsigned int  v8u;

// fp32 -> bf16 with round-to-nearest-even
__device__ __forceinline__ unsigned short f2bf(float f) {
    unsigned int u = __float_as_uint(f);
    u += 0x7fffu + ((u >> 16) & 1u);
    return (unsigned short)(u >> 16);
}
__device__ __forceinline__ unsigned int f2bf2(float a, float b) {
    return (unsigned int)f2bf(a) | ((unsigned int)f2bf(b) << 16);
}

// ---------------------------------------------------------------------------
// 1) ngram rolling hash.  seg[t]==seg[t-n]  <=>  no EOS among ids[t-n..t-1].
// ---------------------------------------------------------------------------
__global__ __launch_bounds__(256) void hash_kernel(const int* __restrict__ ids,
                                                   int* __restrict__ nid) {
    int t = blockIdx.x * 256 + threadIdx.x;      // 0 .. M_-1
    if (t >= M_) return;
    int s = t & (S_ - 1);
    long long id0 = ids[t];
    long long p1 = (s >= 1) ? (long long)ids[t - 1] : (long long)EOS_;
    long long p2 = (s >= 2) ? (long long)ids[t - 2] : (long long)EOS_;
    long long sh1 = (p1 != EOS_) ? p1 : 0;                    // shift-by-1
    long long sh2 = (p1 != EOS_ && p2 != EOS_) ? p2 : 0;      // shift-by-2
#pragma unroll
    for (int idx = 0; idx < NEMB_; ++idx) {
        long long dim = (long long)(4 * VOCAB_) + 2 * idx + 1;   // 128001 + 2*idx
        long long ng;
        if (idx < 2) {              // bigram embedders (i == 2)
            ng = id0 + sh1 * (long long)VOCAB_;     // VOCAB % dim == VOCAB
        } else {                    // trigram embedders (i == 3)
            long long m1 = ((long long)VOCAB_ * (long long)VOCAB_) % dim;
            ng = id0 + sh1 * (long long)VOCAB_ + sh2 * m1;
        }
        nid[idx * M_ + t] = (int)(ng % dim);
    }
}

// ---------------------------------------------------------------------------
// 2) proj fp32 -> bf16 staging:  P[idx][h][d], row-major, d contiguous
// ---------------------------------------------------------------------------
__global__ __launch_bounds__(256) void cvt_proj_kernel(const float* __restrict__ p0,
                                                       const float* __restrict__ p1,
                                                       const float* __restrict__ p2,
                                                       const float* __restrict__ p3,
                                                       unsigned short* __restrict__ P) {
    const float* ps[NEMB_] = {p0, p1, p2, p3};
    const int perTblV4 = (HID_ * EDIM_) / 4;                 // 262144 float4 per table
    int tid = blockIdx.x * 256 + threadIdx.x;
    int idx = tid >> 18;                                     // / 262144
    int off = tid & (perTblV4 - 1);
    float4 v = ((const float4*)ps[idx])[off];
    uint2 o;
    o.x = f2bf2(v.x, v.y);
    o.y = f2bf2(v.z, v.w);
    ((uint2*)(P + (size_t)idx * HID_ * EDIM_))[off] = o;
}

// ---------------------------------------------------------------------------
// 3) gather + fp32->bf16:  A[idx][m][d], row-major.  One wave per (idx, m) row.
// ---------------------------------------------------------------------------
__global__ __launch_bounds__(256) void gather_kernel(const int* __restrict__ nid,
                                                     const float* __restrict__ e0,
                                                     const float* __restrict__ e1,
                                                     const float* __restrict__ e2,
                                                     const float* __restrict__ e3,
                                                     unsigned short* __restrict__ A) {
    const float* es[NEMB_] = {e0, e1, e2, e3};
    int wave = blockIdx.x * 8 + (threadIdx.x >> 5);          // 0 .. 4*M_-1
    int lane = threadIdx.x & 31;
    int idx  = wave >> 13;                                   // / 8192
    int m    = wave & (M_ - 1);
    const float* src = es[idx] + (size_t)nid[idx * M_ + m] * EDIM_ + lane * 16;
    unsigned short* dst = A + ((size_t)idx * M_ + m) * EDIM_;
    float4 a = ((const float4*)src)[0];
    float4 b = ((const float4*)src)[1];
    float4 c = ((const float4*)src)[2];
    float4 d = ((const float4*)src)[3];
    uint4 lo, hi;
    lo.x = f2bf2(a.x, a.y); lo.y = f2bf2(a.z, a.w);
    lo.z = f2bf2(b.x, b.y); lo.w = f2bf2(b.z, b.w);
    hi.x = f2bf2(c.x, c.y); hi.y = f2bf2(c.z, c.w);
    hi.z = f2bf2(d.x, d.y); hi.w = f2bf2(d.z, d.w);
    ((uint4*)dst)[lane * 2 + 0] = lo;
    ((uint4*)dst)[lane * 2 + 1] = hi;
}

// ---------------------------------------------------------------------------
// 4) fused GEMM with async global->LDS double buffering.
//    Block: 256 thr (8 waves) -> 256(M) x 128(N) tile, K-step 32.
//    Wave : 64 x 64 = 4x4 grid of v_wmma_f32_16x16x32_bf16 accumulators.
//    LDS  : A panel 256x32, B panel 128x32 (bf16), rows padded to 80 B so the
//           lane-strided ds_load_b128 fragment reads are bank-conflict-free,
//           double buffered = 60 KiB.
//    Copy : GLOBAL_LOAD_ASYNC_TO_LDS_B128 (ASYNCcnt), 6 issues/wave/panel;
//           in-order completion => s_wait_asynccnt 6 ensures panel g landed
//           while panel g+1 is in flight.
// ---------------------------------------------------------------------------
__device__ __forceinline__ v16bf lds_frag(const unsigned short* p, int hiOff16) {
    // p: 16B-aligned LDS pointer; elements 0..7 at p, 8..15 at p + hiOff16 ushorts
    const uint4* q = (const uint4*)p;
    uint4 lo = q[0];
    uint4 hi = q[hiOff16 >> 3];
    v8u r = {lo.x, lo.y, lo.z, lo.w, hi.x, hi.y, hi.z, hi.w};
    return __builtin_bit_cast(v16bf, r);
}

__device__ __forceinline__ void async_cp16(unsigned ldsDst, const unsigned short* src) {
    asm volatile("global_load_async_to_lds_b128 %0, %1, off"
                 :: "v"(ldsDst), "v"(src) : "memory");
}

__global__ __launch_bounds__(256) void gemm_kernel(const unsigned short* __restrict__ A,
                                                   const unsigned short* __restrict__ P,
                                                   const int* __restrict__ ids,
                                                   const float* __restrict__ wemb,
                                                   float* __restrict__ out) {
    __shared__ __attribute__((aligned(16))) unsigned short As[2][BM_][LDSROW_];
    __shared__ __attribute__((aligned(16))) unsigned short Bs[2][BN_][LDSROW_];

    const int t    = threadIdx.x;
    const int lane = t & 31;
    const int w    = t >> 5;
    const int wm   = w & 3;                 // 4 waves along M
    const int wn   = w >> 2;                // 2 waves along N
    const int mB   = blockIdx.y * BM_;      // block M origin
    const int nB   = blockIdx.x * BN_;      // block N origin
    const int l15  = lane & 15;
    const int kHalfA = (lane >> 4) << 3;    // 0 or 8   (ushorts)
    const int kHalfB = (lane >> 4) << 3;    // 0 or 8   (ushorts)

    const unsigned asBase = (unsigned)(unsigned long long)(void*)&As[0][0][0];
    const unsigned bsBase = (unsigned)(unsigned long long)(void*)&Bs[0][0][0];
    const unsigned aBufB  = (unsigned)(BM_ * LDSROW_ * 2);   // bytes per A buffer
    const unsigned bBufB  = (unsigned)(BN_ * LDSROW_ * 2);   // bytes per B buffer

    v8f acc[4][4];
#pragma unroll
    for (int i = 0; i < 4; ++i)
#pragma unroll
        for (int j = 0; j < 4; ++j) acc[i][j] = (v8f)0.0f;

    // ---- async panel copy: iteration g covers (idx = g>>4, kt = g&15) ------
    auto issue_panel = [&](int g, int buf) {
        const int idx = g >> 4, kt = g & 15;
        const unsigned short* Ag = A + ((size_t)idx * M_ + mB) * EDIM_ + kt * BK_;
        const unsigned short* Bg = P + ((size_t)idx * HID_ + nB) * EDIM_ + kt * BK_;
        const unsigned aOff = asBase + (unsigned)buf * aBufB;
        const unsigned bOff = bsBase + (unsigned)buf * bBufB;
#pragma unroll
        for (int q = 0; q < 4; ++q) {            // A: 1024 x 16B chunks / 256 thr
            int c = t + 256 * q, row = c >> 2, c16 = c & 3;
            async_cp16(aOff + (unsigned)(row * (LDSROW_ * 2) + c16 * 16),
                       Ag + (size_t)row * EDIM_ + c16 * 8);
        }
#pragma unroll
        for (int q = 0; q < 2; ++q) {            // B: 512 x 16B chunks / 256 thr
            int c = t + 256 * q, row = c >> 2, c16 = c & 3;
            async_cp16(bOff + (unsigned)(row * (LDSROW_ * 2) + c16 * 16),
                       Bg + (size_t)row * EDIM_ + c16 * 8);
        }
    };

    issue_panel(0, 0);                           // prologue: panel 0 -> buf 0

    for (int g = 0; g < NITER_; ++g) {
        const int buf = g & 1;
        if (g + 1 < NITER_) {
            issue_panel(g + 1, buf ^ 1);         // prefetch next panel
            asm volatile("s_wait_asynccnt 6" ::: "memory");   // panel g landed
        } else {
            asm volatile("s_wait_asynccnt 0" ::: "memory");
        }
        __syncthreads();                         // panel g visible to all waves

        v16bf a[4], b[4];
#pragma unroll
        for (int i = 0; i < 4; ++i)
            a[i] = lds_frag(&As[buf][wm * 64 + i * 16 + l15][kHalfA], 16);
#pragma unroll
        for (int j = 0; j < 4; ++j)
            b[j] = lds_frag(&Bs[buf][wn * 64 + j * 16 + l15][kHalfB], 16);
#pragma unroll
        for (int i = 0; i < 4; ++i)
#pragma unroll
            for (int j = 0; j < 4; ++j)
                acc[i][j] = __builtin_amdgcn_wmma_f32_16x16x32_bf16(
                    false, a[i], false, b[j], (short)0, acc[i][j], false, false);

        __syncthreads();                         // all reads done before overwrite
    }

    // Epilogue: C frag element r -> (M = r + 8*(lane>=16), N = lane&15).
    // Fuse word_emb gather + 1/5 scale; each output written exactly once.
    const int mh = (lane >> 4) << 3;             // 0 or 8
#pragma unroll
    for (int i = 0; i < 4; ++i) {
#pragma unroll
        for (int r = 0; r < 8; ++r) {
            const int gm = mB + wm * 64 + i * 16 + r + mh;
            const float* wrow = wemb + (size_t)ids[gm] * HID_;
            float* orow = out + (size_t)gm * HID_;
#pragma unroll
            for (int j = 0; j < 4; ++j) {
                const int gn = nB + wn * 64 + j * 16 + l15;
                orow[gn] = (wrow[gn] + acc[i][j][r]) * 0.2f;
            }
        }
    }
}

// ---------------------------------------------------------------------------
// launch
// ---------------------------------------------------------------------------
extern "C" void kernel_launch(void* const* d_in, const int* in_sizes, int n_in,
                              void* d_out, int out_size, void* d_ws, size_t ws_size,
                              hipStream_t stream) {
    // setup_inputs() dict order:
    // 0 input_ids, 1 word_emb, 2 emb_0, 3 proj_0, 4 emb_1, 5 proj_1,
    // 6 emb_2, 7 proj_2, 8 emb_3, 9 proj_3
    const int*   ids  = (const int*)d_in[0];
    const float* wemb = (const float*)d_in[1];
    const float* e0 = (const float*)d_in[2];
    const float* p0 = (const float*)d_in[3];
    const float* e1 = (const float*)d_in[4];
    const float* p1 = (const float*)d_in[5];
    const float* e2 = (const float*)d_in[6];
    const float* p2 = (const float*)d_in[7];
    const float* e3 = (const float*)d_in[8];
    const float* p3 = (const float*)d_in[9];
    float* out = (float*)d_out;

    // workspace layout (16B aligned):
    //   A  : bf16 [4][8192][512]  = 32 MiB
    //   P  : bf16 [4][2048][512]  =  8 MiB
    //   nid: int  [4][8192]       = 128 KiB
    char* ws = (char*)d_ws;
    unsigned short* A   = (unsigned short*)ws;
    unsigned short* P   = (unsigned short*)(ws + (size_t)NEMB_ * M_ * EDIM_ * 2);
    int*            nid = (int*)(ws + (size_t)NEMB_ * M_ * EDIM_ * 2
                                    + (size_t)NEMB_ * HID_ * EDIM_ * 2);

    hash_kernel<<<M_ / 256, 256, 0, stream>>>(ids, nid);
    cvt_proj_kernel<<<(NEMB_ * HID_ * EDIM_ / 4) / 256, 256, 0, stream>>>(p0, p1, p2, p3, P);
    gather_kernel<<<(NEMB_ * M_) / 8, 256, 0, stream>>>(nid, e0, e1, e2, e3, A);

    dim3 grid(HID_ / BN_, M_ / BM_);   // (16, 32)
    gemm_kernel<<<grid, 256, 0, stream>>>(A, P, ids, wemb, out);
}
